// SparseSECOND_69595650064515
// MI455X (gfx1250) — compile-verified, matrix operand order checked by reading, not run on hardware
//
#include <hip/hip_runtime.h>
#include <hip/hip_bf16.h>

typedef __attribute__((ext_vector_type(16))) __bf16 v16bf;
typedef __attribute__((ext_vector_type(8)))  float  v8f;

#define BN_EPS 1e-3f

// ---------------------------------------------------------------------------
// Weight pre-pack: HWIO f32 -> bf16 B-fragment layout.
// Packed element index i = (((tap*nChunks+ch)*nCo16 + co16)*32 + lane)*16 + e
// lane = 16*half + n (col N = n); element e -> k = (e<8 ? e : e+8) + 8*half
// (CDNA5 ISA 7.12.2 16-bit 32x16 B layout).
// ---------------------------------------------------------------------------
__global__ void pack_weights_k(const float* __restrict__ W, __bf16* __restrict__ out,
                               int Cin, int Cout, long N)
{
    long i = (long)blockIdx.x * blockDim.x + threadIdx.x;
    if (i >= N) return;
    int  e    = (int)(i & 15);
    int  lane = (int)((i >> 4) & 31);
    long f    = i >> 9;
    int  nCo16 = Cout >> 4;
    int  co16  = (int)(f % nCo16);
    long tc    = f / nCo16;
    int  nCh   = Cin >> 5;
    int  ch    = (int)(tc % nCh);
    int  tap   = (int)(tc / nCh);
    int  half  = lane >> 4;
    int  n     = lane & 15;
    int  k     = (e < 8 ? e : e + 8) + 8 * half;
    int  ci    = ch * 32 + k;
    int  co    = co16 * 16 + n;
    out[i] = (__bf16)W[((long)tap * Cin + ci) * Cout + co];
}

// ---------------------------------------------------------------------------
// Apply input mask and convert f32 -> bf16 (first conv's input).
// ---------------------------------------------------------------------------
__global__ void init_x_k(const float* __restrict__ x, const unsigned char* __restrict__ m,
                         __bf16* __restrict__ out, long N, int C)
{
    long i = (long)blockIdx.x * blockDim.x + threadIdx.x;
    if (i >= N) return;
    out[i] = m[i / C] ? (__bf16)x[i] : (__bf16)0.0f;
}

// ---------------------------------------------------------------------------
// 2x2 "any" mask pooling (ME stride-2 active-site rule).
// ---------------------------------------------------------------------------
__global__ void pool_mask_k(const unsigned char* __restrict__ in, unsigned char* __restrict__ out,
                            int Ho, int Wo, int Hi, int Wi)
{
    int i = blockIdx.x * blockDim.x + threadIdx.x;
    int total = 4 * Ho * Wo;
    if (i >= total) return;
    int xo = i % Wo; int t = i / Wo; int yo = t % Ho; int b = t / Ho;
    const unsigned char* r0 = in + ((long)(b * Hi + 2 * yo) * Wi + 2 * xo);
    const unsigned char* r1 = r0 + Wi;
    out[i] = (unsigned char)((r0[0] | r0[1] | r1[0] | r1[1]) ? 1 : 0);
}

__global__ void count_mask_k(const unsigned char* __restrict__ m, long L, float* __restrict__ cnt)
{
    __shared__ float s[256];
    float acc = 0.0f;
    for (long i = (long)blockIdx.x * 256 + threadIdx.x; i < L; i += (long)gridDim.x * 256)
        acc += m[i] ? 1.0f : 0.0f;
    s[threadIdx.x] = acc;
    __syncthreads();
    for (int o = 128; o > 0; o >>= 1) {
        if ((int)threadIdx.x < o) s[threadIdx.x] += s[threadIdx.x + o];
        __syncthreads();
    }
    if (threadIdx.x == 0) atomicAdd(cnt, s[0]);
}

__global__ void zero_f32_k(float* __restrict__ p, int n)
{
    int i = blockIdx.x * blockDim.x + threadIdx.x;
    if (i < n) p[i] = 0.0f;
}

// ---------------------------------------------------------------------------
// Implicit-GEMM 3x3 conv via v_wmma_f32_16x16x32_bf16, templated on
// (CIN, COUT, STRIDE) so chunk counts / pitches / divisions are compile-time.
//
// Block = 128 threads = 4 wave32s. Block tile: 32 output pixels of one output
// row (MT=2 16-px A-tiles per wave) x full COUT (NPW 16-col tiles per wave).
// Each B-fragment (1KB from L2) is reused by MT WMMAs -> 2x FLOP per L2 byte
// vs a 16-px tile; weight traffic is the binding resource since B-fragments
// cannot be shared across waves (each wave owns distinct co16 tiles).
// Per kh (3 input rows): stage a NPIX x CIN bf16 row segment into LDS once
// (NPIX = 31*STRIDE + 3 covers both M-tiles and all three kw taps), then
// sweep cin-chunks x kw reading A-fragments from LDS.
// LDS row pitch padded by 8 bf16 (16B) -> A-frag rows land on distinct bank
// groups (mod 64), avoiding 16-way conflicts.
// Out-of-range kh rows contribute zero -> block-uniform skip (EXEC stays all
// ones around WMMA). Output multiplied by pooled mask, f32 to Y.
// ---------------------------------------------------------------------------
template<int CIN, int COUT, int STRIDE>
__global__ __launch_bounds__(128)
void conv3x3_wmma(const __bf16* __restrict__ X, const __bf16* __restrict__ Wp,
                  const unsigned char* __restrict__ Mout, float* __restrict__ Y,
                  int Hin, int Win, int Ho, int Wo)
{
    constexpr int MT    = 2;                // 16-px M-tiles per wave
    constexpr int NPW   = COUT / 64;        // 16-col tiles per wave
    constexpr int NCH   = CIN / 32;         // K chunks of 32 channels
    constexpr int NCO16 = COUT / 16;
    constexpr int NPIX  = 31 * STRIDE + 3;  // staged input pixels per row
    constexpr int PITCH = CIN + 8;          // LDS pitch (bf16) with bank pad
    constexpr int CQ    = CIN / 4;          // uint2 (4 bf16) groups per pixel
    constexpr int NU2   = NPIX * CQ;        // total uint2 to stage

    __shared__ __align__(16) __bf16 ldsA[NPIX * PITCH];

    const int b    = blockIdx.z;
    const int oy   = blockIdx.y;
    const int ox0  = blockIdx.x * 32;
    const int tid  = threadIdx.x;
    const int wave = tid >> 5;
    const int lane = tid & 31;
    const int half = lane >> 4;
    const int mn   = lane & 15;

    v8f zero = {};
    v8f acc[MT][NPW];
#pragma unroll
    for (int t = 0; t < MT; ++t)
#pragma unroll
        for (int j = 0; j < NPW; ++j) acc[t][j] = zero;

    const int ixBase = ox0 * STRIDE - 1;    // input x of staged pixel 0

    for (int kh = 0; kh < 3; ++kh) {
        const int iy = oy * STRIDE + kh - 1;
        if (iy >= 0 && iy < Hin) {          // block-uniform: skip zero rows
            // ---- stage NPIX x CIN row segment into LDS (zero-padded) ----
            const __bf16* rowPtr = X + (long)(b * Hin + iy) * Win * CIN;
#pragma unroll 2
            for (int u = tid; u < NU2; u += 128) {
                const int s  = u / CQ;      // CQ is a power of two
                const int c4 = u - s * CQ;
                const int ix = ixBase + s;
                uint2 v; v.x = 0u; v.y = 0u;
                if (ix >= 0 && ix < Win)
                    v = *(const uint2*)(rowPtr + (long)ix * CIN + c4 * 4);
                *(uint2*)(ldsA + s * PITCH + c4 * 4) = v;
            }
            __syncthreads();

            // ---- consume: NCH chunks x 3 kw taps, all from LDS ----
            for (int ch = 0; ch < NCH; ++ch) {
#pragma unroll
                for (int kw = 0; kw < 3; ++kw) {
                    // MT A-fragments from LDS (rows M = mn, pixels offset 16)
                    union { uint4 u[2]; v16bf v; } a[MT];
#pragma unroll
                    for (int t = 0; t < MT; ++t) {
                        const int px = (mn + 16 * t) * STRIDE + kw;
                        const uint4* ap =
                            (const uint4*)(ldsA + px * PITCH + ch * 32 + half * 8);
                        a[t].u[0] = ap[0];  // channels ch*32 + 8*half + (0..7)
                        a[t].u[1] = ap[2];  // channels ch*32 + 8*half + (16..23)
                    }
                    const int tap = kh * 3 + kw;
#pragma unroll
                    for (int j = 0; j < NPW; ++j) {
                        const int  co16  = wave * NPW + j;
                        const long fbase =
                            (((long)tap * NCH + ch) * NCO16 + co16) * 32 + lane;
                        const uint4* wq = (const uint4*)(Wp + fbase * 16);
                        union { uint4 u[2]; v16bf v; } bb;
                        bb.u[0] = wq[0];
                        bb.u[1] = wq[1];
#pragma unroll
                        for (int t = 0; t < MT; ++t)
                            acc[t][j] = __builtin_amdgcn_wmma_f32_16x16x32_bf16(
                                false, a[t].v, false, bb.v, (short)0, acc[t][j],
                                false, false);
                    }
                }
            }
            __syncthreads();
        }
    }

    // Epilogue: D layout — lane(half,mn): N = mn, rows M = r + 8*half
    const long rowOut = (long)(b * Ho + oy) * Wo;
#pragma unroll
    for (int t = 0; t < MT; ++t) {
#pragma unroll
        for (int r = 0; r < 8; ++r) {
            const int m  = r + 8 * half;
            const int ox = ox0 + 16 * t + m;
            if (ox < Wo) {
                const float mv = Mout[rowOut + ox] ? 1.0f : 0.0f;
#pragma unroll
                for (int j = 0; j < NPW; ++j) {
                    const int co = (wave * NPW + j) * 16 + mn;
                    Y[(rowOut + ox) * (long)COUT + co] = acc[t][j][r] * mv;
                }
            }
        }
    }
}

// ---------------------------------------------------------------------------
// Per-channel sum / sumsq over masked conv output (inactive sites are 0).
// blockDim = C; coalesced channel-major reads; atomic accumulation.
// ---------------------------------------------------------------------------
__global__ void bn_stats_k(const float* __restrict__ Y, float* __restrict__ stats,
                           long P, int C, int PPB)
{
    const int c = threadIdx.x;
    long p0 = (long)blockIdx.x * PPB;
    long p1 = p0 + PPB; if (p1 > P) p1 = P;
    float s = 0.0f, q = 0.0f;
    for (long p = p0; p < p1; ++p) {
        float v = Y[p * C + c];
        s += v; q += v * v;
    }
    atomicAdd(&stats[c], s);
    atomicAdd(&stats[C + c], q);
}

// ---------------------------------------------------------------------------
// BN(train, active-site stats) + ReLU + mask. Writes bf16 next-layer input,
// and (at stage ends) the f32 stage output.
// ---------------------------------------------------------------------------
__global__ void bn_apply_k(const float* __restrict__ Y, const unsigned char* __restrict__ m,
                           const float* __restrict__ g, const float* __restrict__ bta,
                           const float* __restrict__ stats, const float* __restrict__ cnt,
                           __bf16* __restrict__ Xn, float* __restrict__ Out, long P, int C)
{
    long i = (long)blockIdx.x * blockDim.x + threadIdx.x;
    if (i >= P * C) return;
    int  c = (int)(i % C);
    long p = i / C;
    float n    = fmaxf(cnt[0], 1.0f);
    float mean = stats[c] / n;
    float var  = stats[C + c] / n - mean * mean;
    float rs   = rsqrtf(var + BN_EPS);
    float y    = (Y[i] - mean) * rs * g[c] + bta[c];
    y = m[p] ? fmaxf(y, 0.0f) : 0.0f;
    Xn[i] = (__bf16)y;
    if (Out) Out[i] = y;
}

// ---------------------------------------------------------------------------
// Host orchestration
// ---------------------------------------------------------------------------
struct LayerDesc { int Hin, Win, Ho, Wo, Cin, Cout, stride, stage, stageEnd; };

static const LayerDesc LD[16] = {
    {496,432,248,216, 64, 64,2,0,0},
    {248,216,248,216, 64, 64,1,0,0},
    {248,216,248,216, 64, 64,1,0,0},
    {248,216,248,216, 64, 64,1,0,1},
    {248,216,124,108, 64,128,2,1,0},
    {124,108,124,108,128,128,1,1,0},
    {124,108,124,108,128,128,1,1,0},
    {124,108,124,108,128,128,1,1,0},
    {124,108,124,108,128,128,1,1,0},
    {124,108,124,108,128,128,1,1,1},
    {124,108, 62, 54,128,256,2,2,0},
    { 62, 54, 62, 54,256,256,1,2,0},
    { 62, 54, 62, 54,256,256,1,2,0},
    { 62, 54, 62, 54,256,256,1,2,0},
    { 62, 54, 62, 54,256,256,1,2,0},
    { 62, 54, 62, 54,256,256,1,2,1},
};

static inline size_t align256(size_t x) { return (x + 255) & ~(size_t)255; }
static inline int ceil_div_l(long a, int b) { return (int)((a + b - 1) / b); }

extern "C" void kernel_launch(void* const* d_in, const int* in_sizes, int n_in,
                              void* d_out, int out_size, void* d_ws, size_t ws_size,
                              hipStream_t stream)
{
    (void)in_sizes; (void)n_in; (void)out_size; (void)ws_size;

    const float*         x0 = (const float*)d_in[0];
    const unsigned char* m0 = (const unsigned char*)d_in[1];   // jax bool -> 1 byte

    // ---- workspace carve-up ----
    char*  ws  = (char*)d_ws;
    size_t off = 0;
    auto carve = [&](size_t bytes) { size_t o = off; off = align256(off + bytes); return o; };

    __bf16* XA   = (__bf16*)(ws + carve((size_t)4 * 496 * 432 * 64 * 2)); // 109.7 MB
    __bf16* XB   = (__bf16*)(ws + carve((size_t)4 * 248 * 216 * 64 * 2)); //  27.4 MB
    float*  Yb   = (float*) (ws + carve((size_t)4 * 248 * 216 * 64 * 4)); //  54.9 MB
    __bf16* Wpk  = (__bf16*)(ws + carve((size_t)4202496 * 2));            //   8.4 MB
    unsigned char* m1 = (unsigned char*)(ws + carve((size_t)4 * 248 * 216));
    unsigned char* m2 = (unsigned char*)(ws + carve((size_t)4 * 124 * 108));
    unsigned char* m3 = (unsigned char*)(ws + carve((size_t)4 * 62 * 54));
    float* stats = (float*)(ws + carve(512 * sizeof(float)));
    float* cnts  = (float*)(ws + carve(16 * sizeof(float)));

    const unsigned char* mstage[3] = { m1, m2, m3 };
    float* outBase[3] = { (float*)d_out,
                          (float*)d_out + (long)4 * 248 * 216 * 64,
                          (float*)d_out + (long)4 * 248 * 216 * 64 + (long)4 * 124 * 108 * 128 };

    // ---- per-call deterministic re-init of reduction buffers ----
    zero_f32_k<<<1, 64, 0, stream>>>(cnts, 16);

    // ---- mask pooling + active counts per stage ----
    pool_mask_k<<<ceil_div_l((long)4 * 248 * 216, 256), 256, 0, stream>>>(m0, m1, 248, 216, 496, 432);
    pool_mask_k<<<ceil_div_l((long)4 * 124 * 108, 256), 256, 0, stream>>>(m1, m2, 124, 108, 248, 216);
    pool_mask_k<<<ceil_div_l((long)4 * 62 * 54, 256), 256, 0, stream>>>(m2, m3, 62, 54, 124, 108);
    count_mask_k<<<128, 256, 0, stream>>>(m1, (long)4 * 248 * 216, cnts + 0);
    count_mask_k<<<128, 256, 0, stream>>>(m2, (long)4 * 124 * 108, cnts + 1);
    count_mask_k<<<64, 256, 0, stream>>>(m3, (long)4 * 62 * 54, cnts + 2);

    // ---- first conv input: mask + bf16 convert ----
    {
        long N0 = (long)4 * 496 * 432 * 64;
        init_x_k<<<ceil_div_l(N0, 256), 256, 0, stream>>>(x0, m0, XA, N0, 64);
    }

    // ---- pack all weights (bf16 B-fragment layout) ----
    long wOffs[16];
    {
        long wo = 0;
        for (int l = 0; l < 16; ++l) {
            const LayerDesc& d = LD[l];
            long Nw = (long)9 * d.Cin * d.Cout;
            wOffs[l] = wo;
            pack_weights_k<<<ceil_div_l(Nw, 256), 256, 0, stream>>>(
                (const float*)d_in[2 + 3 * l], Wpk + wo, d.Cin, d.Cout, Nw);
            wo += Nw;
        }
    }

    // ---- layer chain ----
    __bf16* Xcur = XA;
    __bf16* Xnxt = XB;
    for (int l = 0; l < 16; ++l) {
        const LayerDesc& d = LD[l];
        const unsigned char* mo = mstage[d.stage];

        dim3 cg((d.Wo + 31) / 32, d.Ho, 4);
        if (d.Cin == 64 && d.Cout == 64 && d.stride == 2)
            conv3x3_wmma< 64, 64,2><<<cg,128,0,stream>>>(Xcur, Wpk+wOffs[l], mo, Yb, d.Hin, d.Win, d.Ho, d.Wo);
        else if (d.Cin == 64 && d.Cout == 64)
            conv3x3_wmma< 64, 64,1><<<cg,128,0,stream>>>(Xcur, Wpk+wOffs[l], mo, Yb, d.Hin, d.Win, d.Ho, d.Wo);
        else if (d.Cin == 64 && d.Cout == 128)
            conv3x3_wmma< 64,128,2><<<cg,128,0,stream>>>(Xcur, Wpk+wOffs[l], mo, Yb, d.Hin, d.Win, d.Ho, d.Wo);
        else if (d.Cin == 128 && d.Cout == 128)
            conv3x3_wmma<128,128,1><<<cg,128,0,stream>>>(Xcur, Wpk+wOffs[l], mo, Yb, d.Hin, d.Win, d.Ho, d.Wo);
        else if (d.Cin == 128 && d.Cout == 256)
            conv3x3_wmma<128,256,2><<<cg,128,0,stream>>>(Xcur, Wpk+wOffs[l], mo, Yb, d.Hin, d.Win, d.Ho, d.Wo);
        else
            conv3x3_wmma<256,256,1><<<cg,128,0,stream>>>(Xcur, Wpk+wOffs[l], mo, Yb, d.Hin, d.Win, d.Ho, d.Wo);

        const long P = (long)4 * d.Ho * d.Wo;
        zero_f32_k<<<2, 256, 0, stream>>>(stats, 2 * d.Cout);
        bn_stats_k<<<ceil_div_l(P, 128), d.Cout, 0, stream>>>(Yb, stats, P, d.Cout, 128);

        const float* g = (const float*)d_in[3 + 3 * l];
        const float* b = (const float*)d_in[4 + 3 * l];
        float* outPtr = d.stageEnd ? outBase[d.stage] : nullptr;
        bn_apply_k<<<ceil_div_l(P * d.Cout, 256), 256, 0, stream>>>(
            Yb, mo, g, b, stats, cnts + d.stage, Xnxt, outPtr, P, d.Cout);

        __bf16* t = Xcur; Xcur = Xnxt; Xnxt = t;
    }
}
